// MultiHeadAttention_55113020342672
// MI455X (gfx1250) — matrix operand order repaired and checked by reading, not compiled, not run on hardware
//
#include <hip/hip_runtime.h>

// ---------------------------------------------------------------------------
// MHA forward for MI455X (gfx1250, wave32, WMMA bf16 16x16x32, f32 accum)
// B=4, N=2048, D_IN=D_OUT=1024, H=16, HD=64
// ---------------------------------------------------------------------------

typedef __attribute__((ext_vector_type(16))) __bf16        v16bf;
typedef __attribute__((ext_vector_type(8)))  float         v8f;
typedef __attribute__((ext_vector_type(4)))  unsigned int  uint4v;
typedef __attribute__((ext_vector_type(4)))  int           v4i;

union FragA { v16bf v; uint4v q[2]; };

#define WMMA_BF16(a, b, c) \
  __builtin_amdgcn_wmma_f32_16x16x32_bf16(false, (a), false, (b), (short)0, (c), false, false)

#define NB   4
#define NN   2048
#define ND   1024
#define NH   16
#define HD   64
#define MROWS (NB * NN)      // 8192

// ---------------- CDNA5 async global->LDS copy (guarded) --------------------
#if defined(__gfx1250__) && __has_builtin(__builtin_amdgcn_global_load_async_to_lds_b128)
#define HAVE_ASYNC_LDS 1
#else
#define HAVE_ASYNC_LDS 0
#endif

__device__ __forceinline__ void async_copy16(__bf16* l, const __bf16* g) {
#if HAVE_ASYNC_LDS
  __builtin_amdgcn_global_load_async_to_lds_b128(
      (__attribute__((address_space(1))) v4i*)g,
      (__attribute__((address_space(3))) v4i*)l, 0, 0);
#else
  *(uint4v*)l = *(const uint4v*)g;
#endif
}

__device__ __forceinline__ void async_wait_all() {
#if HAVE_ASYNC_LDS
#if __has_builtin(__builtin_amdgcn_s_wait_asynccnt)
  __builtin_amdgcn_s_wait_asynccnt(0);
#else
  asm volatile("s_wait_asynccnt 0" ::: "memory");
#endif
#endif
}

// --------------------------- elementwise converts ---------------------------

__global__ void cvt_f32_to_bf16(const float* __restrict__ in,
                                __bf16* __restrict__ out, int n) {
  int i = blockIdx.x * blockDim.x + threadIdx.x;
  if (i < n) out[i] = (__bf16)in[i];
}

// in: [1024][1024] f32 row-major; out bf16 transposed (out[n][k] = in[k][n])
__global__ void transpose_cvt_bf16(const float* __restrict__ in,
                                   __bf16* __restrict__ out) {
  int i = blockIdx.x * blockDim.x + threadIdx.x;   // i = k*1024 + n
  int k = i >> 10, n = i & 1023;
  out[n * 1024 + k] = (__bf16)in[i];
}

// --------------------------- GEMM helpers -----------------------------------
// One wave computes a 32x64 tile; K-loop double-buffered (t0/t1) so loads for
// the next 32-deep slice overlap the current slice's 8 WMMAs.

struct BTile { FragA a0, a1, b[4]; };

__device__ __forceinline__ void load_tile(BTile& t,
                                          const __bf16* arow0, const __bf16* arow1,
                                          const __bf16* Wt, int n0, int ks,
                                          int half, int l16) {
  t.a0.q[0] = *(const uint4v*)(arow0 + ks + half * 8);
  t.a0.q[1] = *(const uint4v*)(arow0 + ks + 16 + half * 8);
  t.a1.q[0] = *(const uint4v*)(arow1 + ks + half * 8);
  t.a1.q[1] = *(const uint4v*)(arow1 + ks + 16 + half * 8);
#pragma unroll
  for (int f = 0; f < 4; ++f) {
    const __bf16* bp = Wt + (size_t)(n0 + f * 16 + l16) * ND + ks + half * 8;
    t.b[f].q[0] = *(const uint4v*)(bp);
    t.b[f].q[1] = *(const uint4v*)(bp + 16);
  }
}

__device__ __forceinline__ void mma_tile(v8f (*acc)[4], const BTile& t) {
#pragma unroll
  for (int f = 0; f < 4; ++f) {
    acc[0][f] = WMMA_BF16(t.a0.v, t.b[f].v, acc[0][f]);
    acc[1][f] = WMMA_BF16(t.a1.v, t.b[f].v, acc[1][f]);
  }
}

// --------------------------- fused QKV projection ---------------------------
// blockIdx.y selects Q/K/V. Q,K row-major bf16 [8192][1024]; V head-transposed
// Vt[((b*16+h)*64+d)*2048 + n].
__global__ void gemm_qkv(const __bf16* __restrict__ X,
                         const __bf16* __restrict__ WqT,
                         const __bf16* __restrict__ WkT,
                         const __bf16* __restrict__ WvT,
                         __bf16* __restrict__ Qo,
                         __bf16* __restrict__ Ko,
                         __bf16* __restrict__ Vt) {
  const int lane = threadIdx.x & 31;
  const int wid  = (blockIdx.x * blockDim.x + threadIdx.x) >> 5;
  const int NT   = ND / 64;                 // 16 col-tiles
  const int nt   = wid % NT, mt = wid / NT; // mt in [0,256)
  const int which = blockIdx.y;             // 0=Q 1=K 2=V
  const __bf16* Wt = (which == 0) ? WqT : ((which == 1) ? WkT : WvT);
  const int m0 = mt * 32, n0 = nt * 64;
  const int half = lane >> 4, l16 = lane & 15;

  v8f acc[2][4] = {};
  const __bf16* arow0 = X + (size_t)(m0 + l16) * ND;
  const __bf16* arow1 = arow0 + (size_t)16 * ND;

  BTile t0, t1;
  load_tile(t0, arow0, arow1, Wt, n0, 0, half, l16);
  for (int ks = 0; ks < ND; ks += 64) {
    load_tile(t1, arow0, arow1, Wt, n0, ks + 32, half, l16);
    mma_tile(acc, t0);
    if (ks + 64 < ND) load_tile(t0, arow0, arow1, Wt, n0, ks + 64, half, l16);
    mma_tile(acc, t1);
  }

  if (which == 2) {
#pragma unroll
    for (int rb = 0; rb < 2; ++rb) {
#pragma unroll
      for (int f = 0; f < 4; ++f) {
        int c = n0 + f * 16 + l16;
        int h = c >> 6, d = c & 63;
#pragma unroll
        for (int v = 0; v < 8; ++v) {
          int r = m0 + rb * 16 + v + half * 8;
          int bI = r >> 11, n = r & 2047;
          Vt[(size_t)((bI * NH + h) * HD + d) * NN + n] = (__bf16)acc[rb][f][v];
        }
      }
    }
  } else {
    __bf16* O = (which == 0) ? Qo : Ko;
#pragma unroll
    for (int rb = 0; rb < 2; ++rb) {
#pragma unroll
      for (int f = 0; f < 4; ++f) {
        int c = n0 + f * 16 + l16;
#pragma unroll
        for (int v = 0; v < 8; ++v) {
          int r = m0 + rb * 16 + v + half * 8;
          O[(size_t)r * ND + c] = (__bf16)acc[rb][f][v];
        }
      }
    }
  }
}

// --------------------------- flash attention --------------------------------
// One wave per (b, h, 16-query tile). All 4 waves of a block share (b,h), so
// the block cooperatively stages the 32-key K tile and V tile in LDS via
// async global->LDS copies, double-buffered. Online softmax; P bounced through
// per-wave LDS to re-layout C->A fragments.
__global__ void attn_flash(const __bf16* __restrict__ Qb,
                           const __bf16* __restrict__ Kb,
                           const __bf16* __restrict__ Vt,
                           __bf16* __restrict__ AO) {
  __shared__ __align__(16) __bf16 ksh[2][32 * 64];   // [key][d]
  __shared__ __align__(16) __bf16 vsh[2][64 * 32];   // [d][key]
  __shared__ __align__(16) __bf16 pbuf[4][16 * 32];

  const int tid   = threadIdx.x;
  const int lane  = tid & 31;
  const int wslot = tid >> 5;
  const int wid   = (blockIdx.x * blockDim.x + tid) >> 5;
  const int qt = wid & 127;           // query tile (N/16 = 128)
  const int bh = wid >> 7;            // 0..63 (same for all waves in a block)
  const int h = bh & 15, b = bh >> 4;
  const int half = lane >> 4, l16 = lane & 15;

  const __bf16* Qp   = Qb + (size_t)(b * NN + qt * 16 + l16) * ND + h * HD;
  const __bf16* Kblk = Kb + (size_t)(b * NN) * ND + h * HD;          // + row*ND
  const __bf16* Vblk = Vt + (size_t)((b * NH + h) * HD) * NN;        // + d*NN
  __bf16* pb = &pbuf[wslot][0];

  // Cooperative fill of one double-buffer slot for key-block kb (32 keys).
  // K: 32 rows x 128B = 256 x 16B chunks. V: 64 rows x 64B = 256 x 16B chunks.
  auto fill = [&](int buf, int kb) {
#pragma unroll
    for (int c = 0; c < 2; ++c) {
      int ch = tid + c * 128;
      int row = ch >> 3, seg = ch & 7;
      async_copy16(&ksh[buf][row * 64 + seg * 8],
                   Kblk + (size_t)(kb + row) * ND + seg * 8);
      int d = ch >> 2, sg = ch & 3;
      async_copy16(&vsh[buf][d * 32 + sg * 8],
                   Vblk + (size_t)d * NN + kb + sg * 8);
    }
  };

  // Q A-fragments stay in registers: k-slices [0,32) and [32,64)
  FragA aq0, aq1;
  aq0.q[0] = *(const uint4v*)(Qp + half * 8);
  aq0.q[1] = *(const uint4v*)(Qp + 16 + half * 8);
  aq1.q[0] = *(const uint4v*)(Qp + 32 + half * 8);
  aq1.q[1] = *(const uint4v*)(Qp + 48 + half * 8);

  v8f o[4] = {};
  float mi[8], li[8];
#pragma unroll
  for (int v = 0; v < 8; ++v) { mi[v] = -3.0e38f; li[v] = 0.0f; }

  fill(0, 0);
  async_wait_all();
  __syncthreads();

  for (int kb = 0; kb < NN; kb += 32) {
    const int cur = (kb >> 5) & 1;
    if (kb + 32 < NN) fill(cur ^ 1, kb + 32);   // overlap with compute below
    const __bf16* kc = &ksh[cur][0];
    const __bf16* vc = &vsh[cur][0];

    // ---- S = Q K^T for 32 keys: two 16x16 f32 tiles (fragments from LDS) ----
    v8f z = {};
    v8f s0, s1;
    {
      const __bf16* kp = kc + l16 * 64 + half * 8;          // keys 0..15
      FragA b0, b1;
      b0.q[0] = *(const uint4v*)(kp);
      b0.q[1] = *(const uint4v*)(kp + 16);
      b1.q[0] = *(const uint4v*)(kp + 32);
      b1.q[1] = *(const uint4v*)(kp + 48);
      s0 = WMMA_BF16(aq0.v, b0.v, z);
      s0 = WMMA_BF16(aq1.v, b1.v, s0);
    }
    {
      const __bf16* kp = kc + (16 + l16) * 64 + half * 8;   // keys 16..31
      FragA b0, b1;
      b0.q[0] = *(const uint4v*)(kp);
      b0.q[1] = *(const uint4v*)(kp + 16);
      b1.q[0] = *(const uint4v*)(kp + 32);
      b1.q[1] = *(const uint4v*)(kp + 48);
      s1 = WMMA_BF16(aq0.v, b0.v, z);
      s1 = WMMA_BF16(aq1.v, b1.v, s1);
    }

    // ---- hoist V B-fragments (retire during softmax VALU) ----
    FragA bv[4];
#pragma unroll
    for (int f = 0; f < 4; ++f) {
      const __bf16* vp = vc + (f * 16 + l16) * 32 + half * 8;
      bv[f].q[0] = *(const uint4v*)(vp);
      bv[f].q[1] = *(const uint4v*)(vp + 16);
    }

    // ---- online softmax (row = v + 8*half; its 16 cols live in a half-wave) ----
    float esc[8];
#pragma unroll
    for (int v = 0; v < 8; ++v) {
      float a0 = s0[v] * 0.125f;   // 1/sqrt(64)
      float a1 = s1[v] * 0.125f;
      float t = fmaxf(a0, a1);
      t = fmaxf(t, __shfl_xor(t, 1, 32));
      t = fmaxf(t, __shfl_xor(t, 2, 32));
      t = fmaxf(t, __shfl_xor(t, 4, 32));
      t = fmaxf(t, __shfl_xor(t, 8, 32));
      float mn = fmaxf(mi[v], t);
      float e0 = __expf(a0 - mn);
      float e1 = __expf(a1 - mn);
      s0[v] = e0; s1[v] = e1;
      float r = e0 + e1;
      r += __shfl_xor(r, 1, 32);
      r += __shfl_xor(r, 2, 32);
      r += __shfl_xor(r, 4, 32);
      r += __shfl_xor(r, 8, 32);
      float sc = __expf(mi[v] - mn);
      li[v] = li[v] * sc + r;
      mi[v] = mn;
      esc[v] = sc;
    }
#pragma unroll
    for (int v = 0; v < 8; ++v) {
      o[0][v] *= esc[v]; o[1][v] *= esc[v];
      o[2][v] *= esc[v]; o[3][v] *= esc[v];
    }

    // ---- P (C-layout) -> per-wave LDS row-major [16][32] bf16 ----
#pragma unroll
    for (int v = 0; v < 8; ++v) {
      int m = v + half * 8;
      pb[m * 32 + l16]      = (__bf16)s0[v];
      pb[m * 32 + 16 + l16] = (__bf16)s1[v];
    }
    asm volatile("s_wait_dscnt 0" ::: "memory");   // wave-synchronous reuse

    // ---- P as A-fragment (16x32) ----
    FragA ap;
    const __bf16* pr = pb + l16 * 32;
    ap.q[0] = *(const uint4v*)(pr + half * 8);
    ap.q[1] = *(const uint4v*)(pr + 16 + half * 8);

    // ---- O += P * V ----
#pragma unroll
    for (int f = 0; f < 4; ++f) o[f] = WMMA_BF16(ap.v, bv[f].v, o[f]);

    // next buffer ready + all waves done with this one
    async_wait_all();
    __syncthreads();
  }

  // ---- normalize and store bf16 [b][n][h*64 + c] ----
  float inv[8];
#pragma unroll
  for (int v = 0; v < 8; ++v) inv[v] = 1.0f / li[v];
#pragma unroll
  for (int f = 0; f < 4; ++f) {
    int c = h * HD + f * 16 + l16;
#pragma unroll
    for (int v = 0; v < 8; ++v) {
      int r = b * NN + qt * 16 + v + half * 8;
      AO[(size_t)r * ND + c] = (__bf16)(o[f][v] * inv[v]);
    }
  }
}

// --------------------------- output projection ------------------------------
__global__ void gemm_out(const __bf16* __restrict__ X,     // [8192][1024] bf16
                         const __bf16* __restrict__ WoT,   // [1024][1024] bf16 (col-major)
                         const float* __restrict__ bias,
                         float* __restrict__ Out) {
  const int lane = threadIdx.x & 31;
  const int wid  = (blockIdx.x * blockDim.x + threadIdx.x) >> 5;
  const int NT   = ND / 64;
  const int nt   = wid % NT, mt = wid / NT;
  const int m0 = mt * 32, n0 = nt * 64;
  const int half = lane >> 4, l16 = lane & 15;

  v8f acc[2][4] = {};
  const __bf16* arow0 = X + (size_t)(m0 + l16) * ND;
  const __bf16* arow1 = arow0 + (size_t)16 * ND;

  BTile t0, t1;
  load_tile(t0, arow0, arow1, WoT, n0, 0, half, l16);
  for (int ks = 0; ks < ND; ks += 64) {
    load_tile(t1, arow0, arow1, WoT, n0, ks + 32, half, l16);
    mma_tile(acc, t0);
    if (ks + 64 < ND) load_tile(t0, arow0, arow1, WoT, n0, ks + 64, half, l16);
    mma_tile(acc, t1);
  }

#pragma unroll
  for (int rb = 0; rb < 2; ++rb) {
#pragma unroll
    for (int f = 0; f < 4; ++f) {
      int c = n0 + f * 16 + l16;
      float bo = bias[c];
#pragma unroll
      for (int v = 0; v < 8; ++v) {
        int r = m0 + rb * 16 + v + half * 8;
        Out[(size_t)r * ND + c] = acc[rb][f][v] + bo;
      }
    }
  }
}

// ---------------------------------------------------------------------------

extern "C" void kernel_launch(void* const* d_in, const int* in_sizes, int n_in,
                              void* d_out, int out_size, void* d_ws, size_t ws_size,
                              hipStream_t stream) {
  (void)in_sizes; (void)n_in; (void)out_size; (void)ws_size;

  const float* x   = (const float*)d_in[0];
  const float* w_q = (const float*)d_in[1];
  const float* w_k = (const float*)d_in[2];
  const float* w_v = (const float*)d_in[3];
  const float* w_o = (const float*)d_in[4];
  const float* b_o = (const float*)d_in[5];
  float* out = (float*)d_out;

  char* ws = (char*)d_ws;
  size_t off = 0;
  auto carve = [&](size_t bytes) { void* p = ws + off; off += (bytes + 255) & ~(size_t)255; return p; };
  __bf16* xb  = (__bf16*)carve((size_t)MROWS * ND * 2);   // 16 MB
  __bf16* wqT = (__bf16*)carve((size_t)ND * ND * 2);      //  2 MB
  __bf16* wkT = (__bf16*)carve((size_t)ND * ND * 2);
  __bf16* wvT = (__bf16*)carve((size_t)ND * ND * 2);
  __bf16* woT = (__bf16*)carve((size_t)ND * ND * 2);
  __bf16* Qb  = (__bf16*)carve((size_t)MROWS * ND * 2);   // 16 MB
  __bf16* Kb  = (__bf16*)carve((size_t)MROWS * ND * 2);   // 16 MB
  __bf16* Vt  = (__bf16*)carve((size_t)MROWS * ND * 2);   // 16 MB (head-transposed)
  __bf16* AO  = (__bf16*)carve((size_t)MROWS * ND * 2);   // 16 MB

  // 1) converts
  {
    int n = MROWS * ND;
    cvt_f32_to_bf16<<<n / 256, 256, 0, stream>>>(x, xb, n);
    transpose_cvt_bf16<<<(ND * ND) / 256, 256, 0, stream>>>(w_q, wqT);
    transpose_cvt_bf16<<<(ND * ND) / 256, 256, 0, stream>>>(w_k, wkT);
    transpose_cvt_bf16<<<(ND * ND) / 256, 256, 0, stream>>>(w_v, wvT);
    transpose_cvt_bf16<<<(ND * ND) / 256, 256, 0, stream>>>(w_o, woT);
  }

  // 2) fused QKV projection: (8192/32)*(1024/64) = 4096 waves per matrix
  {
    dim3 grid(1024, 3);
    gemm_qkv<<<grid, 128, 0, stream>>>(xb, wqT, wkT, wvT, Qb, Kb, Vt);
  }

  // 3) flash attention: B*H*(N/16) = 8192 waves, 4 waves/block share (b,h)
  attn_flash<<<2048, 128, 0, stream>>>(Qb, Kb, Vt, AO);

  // 4) output projection + bias -> f32 d_out
  gemm_out<<<1024, 128, 0, stream>>>(AO, woT, b_o, out);
}
